// Multihead_attention_40303973106442
// MI455X (gfx1250) — compile-verified, matrix operand order checked by reading, not run on hardware
//
#include <hip/hip_runtime.h>

#define B_ 4
#define S_ 2048
#define D_ 1024
#define H_ 16
#define HD 64
#define KROW 66  // padded LDS row stride in bf16 (132 B = 33 dwords, odd -> no bank conflicts)

typedef __attribute__((ext_vector_type(16))) __bf16 v16bf;
typedef __attribute__((ext_vector_type(8)))  float  v8f;

// ---------------------------------------------------------------------------
// WMMA helpers (CDNA5 / gfx1250, wave32)
// ---------------------------------------------------------------------------

static __device__ __forceinline__ v8f wmma_bf16(v16bf a, v16bf b, v8f c) {
  // emits v_wmma_f32_16x16x32_bf16
  return __builtin_amdgcn_wmma_f32_16x16x32_bf16(
      /*neg_a=*/false, a, /*neg_b=*/false, b,
      /*c_mod=*/(short)0, c, /*reuse_a=*/false, /*reuse_b=*/false);
}

// Load a 16x32 bf16 tile (row-major, leading dim `ld` in elements) into the
// ISA 16-bit A-matrix register layout (ISA 7.12.2):
//   lanes 0-15  : rows M=0..15, VGPR v<4 holds K=2v,2v+1 ; v>=4 holds K=16+2(v-4),..
//   lanes 16-31 : same rows, K offsets +8 (K=8..15 / K=24..31)
// The same pattern applied to the transposed matrix yields the B operand.
static __device__ __forceinline__ v16bf load_a16(const __bf16* __restrict__ p,
                                                 int ld) {
  const int l  = threadIdx.x & 31;
  const int m  = l & 15;
  const int hi = (l >> 4) & 1;
  const __bf16* row = p + (size_t)m * ld + hi * 8;
  v16bf a;
#pragma unroll
  for (int v = 0; v < 8; ++v) {
    const int kb = (v < 4) ? (2 * v) : (16 + 2 * (v - 4));
    a[2 * v]     = row[kb];
    a[2 * v + 1] = row[kb + 1];
  }
  return a;
}

// Same register layout, reading a padded row already resolved by the caller
// (used for LDS-resident tiles; lowers to ds_load_b128 pairs).
static __device__ __forceinline__ v16bf load_b_row(const __bf16* row) {
  v16bf a;
#pragma unroll
  for (int v = 0; v < 8; ++v) {
    const int kb = (v < 4) ? (2 * v) : (16 + 2 * (v - 4));
    a[2 * v]     = row[kb];
    a[2 * v + 1] = row[kb + 1];
  }
  return a;
}

// ---------------------------------------------------------------------------
// gfx1250 async copy: global -> LDS, 16 B per lane, tracked by ASYNCcnt.
// vdst VGPR carries the per-lane LDS byte address, vaddr the 64-bit global
// address (GV mode, saddr = off).
// ---------------------------------------------------------------------------
static __device__ __forceinline__ void async_cp16(unsigned int lds,
                                                  unsigned long long gaddr) {
  asm volatile("global_load_async_to_lds_b128 %0, %1, off"
               :: "v"(lds), "v"(gaddr)
               : "memory");
}
static __device__ __forceinline__ void wait_async0() {
  asm volatile("s_wait_asynccnt 0x0" ::: "memory");
}
static __device__ __forceinline__ void wait_async4() {
  asm volatile("s_wait_asynccnt 0x4" ::: "memory");
}

// ---------------------------------------------------------------------------
// Kernel 1a: f32 -> bf16 elementwise convert (for x)
// ---------------------------------------------------------------------------
__global__ void k_cvt_bf16(const float* __restrict__ src,
                           __bf16* __restrict__ dst, int nelem) {
  int i = blockIdx.x * blockDim.x + threadIdx.x;
  if (i < nelem) dst[i] = (__bf16)src[i];
}

// ---------------------------------------------------------------------------
// Kernel 1b: transpose + convert the 4 weight matrices (D x D) to bf16.
// dst[n*D + k] = src[k*D + n]  => rows of dst are B^T rows for WMMA B loads.
// ---------------------------------------------------------------------------
__global__ void k_transpose_cvt(const float* __restrict__ wq,
                                const float* __restrict__ wk,
                                const float* __restrict__ wv,
                                const float* __restrict__ wo,
                                __bf16* __restrict__ wqt,
                                __bf16* __restrict__ wkt,
                                __bf16* __restrict__ wvt,
                                __bf16* __restrict__ wot) {
  const int z = blockIdx.z;
  const float* src = (z == 0) ? wq : (z == 1) ? wk : (z == 2) ? wv : wo;
  __bf16* dst      = (z == 0) ? wqt : (z == 1) ? wkt : (z == 2) ? wvt : wot;
  int i = blockIdx.x * blockDim.x + threadIdx.x;  // 0 .. D*D-1
  int nn = i >> 10;
  int kk = i & (D_ - 1);
  dst[i] = (__bf16)src[(size_t)kk * D_ + nn];
}

// ---------------------------------------------------------------------------
// Kernel 2: fused QKV projection GEMM.  M = B*S = 8192, N = D, K = D.
// One wave computes a 32x64 output tile (2 A-tiles x 4 B-tiles -> 8 WMMAs per
// K-step; 0.75 tile-loads per WMMA).  gridDim.z selects q/k/v.
// Q,K stored [B,H,S,64] bf16 ; V stored transposed [B,H,64,S] bf16.
// ---------------------------------------------------------------------------
__global__ void __launch_bounds__(32)
k_qkv_gemm(const __bf16* __restrict__ xb, const __bf16* __restrict__ wqt,
           const __bf16* __restrict__ wkt, const __bf16* __restrict__ wvt,
           const float* __restrict__ bq, const float* __restrict__ bk,
           const float* __restrict__ bv, __bf16* __restrict__ Q,
           __bf16* __restrict__ Kq, __bf16* __restrict__ Vt) {
  const int r0    = blockIdx.x * 32;
  const int n0    = blockIdx.y * 64;
  const int which = blockIdx.z;
  const __bf16* W    = (which == 0) ? wqt : (which == 1) ? wkt : wvt;
  const float*  bias = (which == 0) ? bq : (which == 1) ? bk : bv;

  v8f acc[2][4] = {{{}, {}, {}, {}}, {{}, {}, {}, {}}};
  for (int kk = 0; kk < D_ / 32; ++kk) {
    v16bf ax0 = load_a16(xb + (size_t)r0 * D_ + kk * 32, D_);
    v16bf ax1 = load_a16(xb + (size_t)(r0 + 16) * D_ + kk * 32, D_);
#pragma unroll
    for (int nt = 0; nt < 4; ++nt) {
      v16bf bw = load_a16(W + (size_t)(n0 + nt * 16) * D_ + kk * 32, D_);
      acc[0][nt] = wmma_bf16(ax0, bw, acc[0][nt]);
      acc[1][nt] = wmma_bf16(ax1, bw, acc[1][nt]);
    }
  }

  const int l  = threadIdx.x & 31;
  const int n  = l & 15;
  const int hi = (l >> 4) & 1;
#pragma unroll
  for (int mt = 0; mt < 2; ++mt) {
#pragma unroll
    for (int nt = 0; nt < 4; ++nt) {
#pragma unroll
      for (int v = 0; v < 8; ++v) {
        const int r   = r0 + mt * 16 + v + hi * 8;
        const int j   = n0 + nt * 16 + n;
        const float val = acc[mt][nt][v] + bias[j];
        const int bb = r / S_, s = r % S_;
        const int hh = j >> 6, dd = j & (HD - 1);
        if (which < 2) {
          __bf16* dst = (which == 0) ? Q : Kq;
          dst[((size_t)(bb * H_ + hh) * S_ + s) * HD + dd] = (__bf16)val;
        } else {
          Vt[((size_t)(bb * H_ + hh) * HD + dd) * S_ + s] = (__bf16)val;
        }
      }
    }
  }
}

// ---------------------------------------------------------------------------
// Kernel 3: attention core. One wave per (b, h, 16-row q tile).
// K tiles are streamed through a double-buffered LDS pipeline filled by
// gfx1250 async copies (ASYNCcnt), with a 4 B row pad so the ds_load_b128
// B-operand reads are bank-conflict-free.
// Pass 1: per-LANE online softmax stats, single cross-lane (max,sum) merge.
// Pass 2: recompute scores, write normalized probabilities to attn output,
//         accumulate P @ V via WMMA (P re-laid-out C->A through LDS).
// ---------------------------------------------------------------------------
__global__ void __launch_bounds__(32)
k_attention(const __bf16* __restrict__ Q, const __bf16* __restrict__ Km,
            const __bf16* __restrict__ Vt, float* __restrict__ attn,
            __bf16* __restrict__ ctx) {
  const int qt = blockIdx.x;  // 0..127
  const int h  = blockIdx.y;  // 0..15
  const int b  = blockIdx.z;  // 0..3
  const int m0 = qt * 16;
  const int l  = threadIdx.x & 31;
  const int n  = l & 15;
  const int hi = (l >> 4) & 1;

  const __bf16* Qb = Q  + ((size_t)(b * H_ + h) * S_) * HD;
  const __bf16* Kb = Km + ((size_t)(b * H_ + h) * S_) * HD;
  const __bf16* Vb = Vt + ((size_t)(b * H_ + h) * HD) * S_;
  float* attn_b    = attn + (size_t)(b * H_ + h) * S_ * S_;

  const v16bf aq0 = load_a16(Qb + (size_t)m0 * HD, HD);       // K=0..31 of hd
  const v16bf aq1 = load_a16(Qb + (size_t)m0 * HD + 32, HD);  // K=32..63

  const float scale = 0.125f;  // 1/sqrt(64)

  // ---- async K-tile pipeline setup --------------------------------------
  __shared__ __bf16 kbuf[2][16 * KROW];  // 2 x 2112 B padded K tiles
  __shared__ __bf16 pbuf[16 * 32];       // 1 KB P staging (C-layout -> A-layout)

  unsigned int lbase[2];
  lbase[0] = (unsigned int)(uintptr_t)&kbuf[0][0];
  lbase[1] = (unsigned int)(uintptr_t)&kbuf[1][0];
  unsigned int loff[4];
#pragma unroll
  for (int q = 0; q < 4; ++q) {
    const int t = (q * 32 + l) * 16;  // byte index within the 2048 B tile
    loff[q] = (unsigned int)((t >> 7) * (KROW * 2) + (t & 127));
  }
  const unsigned long long gK = (unsigned long long)(uintptr_t)Kb;
  const unsigned long long gl = (unsigned long long)(l * 16);

  // ---- pass 1: per-lane online (max, sum) -------------------------------
  float lmax[8], lsum[8];
#pragma unroll
  for (int v = 0; v < 8; ++v) { lmax[v] = -3.0e38f; lsum[v] = 0.0f; }

  // prologue: tile 0 -> buffer 0
#pragma unroll
  for (int q = 0; q < 4; ++q)
    async_cp16(lbase[0] + loff[q], gK + gl + (unsigned long long)(q * 512));

  for (int kt = 0; kt < S_ / 16; ++kt) {
    const int cur = kt & 1;
    if (kt + 1 < S_ / 16) {
      const unsigned long long gt = gK + (unsigned long long)(kt + 1) * 2048ull + gl;
#pragma unroll
      for (int q = 0; q < 4; ++q)
        async_cp16(lbase[cur ^ 1] + loff[q], gt + (unsigned long long)(q * 512));
      wait_async4();  // current tile's 4 copies retired (in-order)
    } else {
      wait_async0();
    }
    v16bf bk0 = load_b_row(&kbuf[cur][(l & 15) * KROW + hi * 8]);
    v16bf bk1 = load_b_row(&kbuf[cur][(l & 15) * KROW + 32 + hi * 8]);
    v8f c = {};
    c = wmma_bf16(aq0, bk0, c);
    c = wmma_bf16(aq1, bk1, c);
#pragma unroll
    for (int v = 0; v < 8; ++v) {
      float t  = c[v] * scale;
      float nm = fmaxf(lmax[v], t);
      // v_exp_f32 are TRANS ops; co-execute with the XDL WMMA pipeline
      lsum[v] = lsum[v] * __expf(lmax[v] - nm) + __expf(t - nm);
      lmax[v] = nm;
    }
  }

  // single cross-lane merge of (max, sum) pairs over the 16 lanes per half
  float rmax[8], rinv[8];
#pragma unroll
  for (int v = 0; v < 8; ++v) {
    float m = lmax[v], s = lsum[v];
#pragma unroll
    for (int off = 1; off < 16; off <<= 1) {
      float om = __shfl_xor(m, off, 32);
      float os = __shfl_xor(s, off, 32);
      float nm = fmaxf(m, om);
      s = s * __expf(m - nm) + os * __expf(om - nm);
      m = nm;
    }
    rmax[v] = m;
    rinv[v] = 1.0f / s;
  }

  // ---- pass 2: probabilities out + P @ V --------------------------------
  v8f acc[4] = {{}, {}, {}, {}};

  // re-prime the async pipeline: tile 0 -> buffer 0
#pragma unroll
  for (int q = 0; q < 4; ++q)
    async_cp16(lbase[0] + loff[q], gK + gl + (unsigned long long)(q * 512));

  for (int kt2 = 0; kt2 < S_ / 32; ++kt2) {
#pragma unroll
    for (int half = 0; half < 2; ++half) {
      const int kt  = kt2 * 2 + half;
      const int cur = kt & 1;
      if (kt + 1 < S_ / 16) {
        const unsigned long long gt =
            gK + (unsigned long long)(kt + 1) * 2048ull + gl;
#pragma unroll
        for (int q = 0; q < 4; ++q)
          async_cp16(lbase[cur ^ 1] + loff[q], gt + (unsigned long long)(q * 512));
        wait_async4();
      } else {
        wait_async0();
      }
      v16bf bk0 = load_b_row(&kbuf[cur][(l & 15) * KROW + hi * 8]);
      v16bf bk1 = load_b_row(&kbuf[cur][(l & 15) * KROW + 32 + hi * 8]);
      v8f c = {};
      c = wmma_bf16(aq0, bk0, c);
      c = wmma_bf16(aq1, bk1, c);
#pragma unroll
      for (int v = 0; v < 8; ++v) {
        const int mr = v + hi * 8;
        float p = __expf(c[v] * scale - rmax[v]) * rinv[v];
        attn_b[(size_t)(m0 + mr) * S_ + kt * 16 + n] = p;  // the big HBM write
        pbuf[mr * 32 + half * 16 + n] = (__bf16)p;         // ds_store (in-order)
      }
    }
    // Re-load P in A-matrix layout from LDS (single wave: DS ops are in-order)
    v16bf ap = load_b_row(&pbuf[(l & 15) * 32 + hi * 8]);
#pragma unroll
    for (int nt = 0; nt < 4; ++nt) {
      // B^T rows come straight from V^T [64 x S], contiguous in the key dim
      v16bf bv = load_a16(Vb + (size_t)(nt * 16) * S_ + kt2 * 32, S_);
      acc[nt] = wmma_bf16(ap, bv, acc[nt]);
    }
  }

  // store context tile (bf16) into ctx [B*S, D]
#pragma unroll
  for (int nt = 0; nt < 4; ++nt) {
#pragma unroll
    for (int v = 0; v < 8; ++v) {
      const int mr = v + hi * 8;
      ctx[(size_t)(b * S_ + m0 + mr) * D_ + h * HD + nt * 16 + n] =
          (__bf16)acc[nt][v];
    }
  }
}

// ---------------------------------------------------------------------------
// Kernel 4: output projection  out = ctx @ wo + bo  (M=8192, N=K=1024), f32 out
// 32x64 tile per wave, same operand-reuse scheme as kernel 2.
// ---------------------------------------------------------------------------
__global__ void __launch_bounds__(32)
k_out_gemm(const __bf16* __restrict__ ctx, const __bf16* __restrict__ wot,
           const float* __restrict__ bo, float* __restrict__ out) {
  const int r0 = blockIdx.x * 32;
  const int n0 = blockIdx.y * 64;
  v8f acc[2][4] = {{{}, {}, {}, {}}, {{}, {}, {}, {}}};
  for (int kk = 0; kk < D_ / 32; ++kk) {
    v16bf ax0 = load_a16(ctx + (size_t)r0 * D_ + kk * 32, D_);
    v16bf ax1 = load_a16(ctx + (size_t)(r0 + 16) * D_ + kk * 32, D_);
#pragma unroll
    for (int nt = 0; nt < 4; ++nt) {
      v16bf bw = load_a16(wot + (size_t)(n0 + nt * 16) * D_ + kk * 32, D_);
      acc[0][nt] = wmma_bf16(ax0, bw, acc[0][nt]);
      acc[1][nt] = wmma_bf16(ax1, bw, acc[1][nt]);
    }
  }
  const int l  = threadIdx.x & 31;
  const int n  = l & 15;
  const int hi = (l >> 4) & 1;
#pragma unroll
  for (int mt = 0; mt < 2; ++mt) {
#pragma unroll
    for (int nt = 0; nt < 4; ++nt) {
#pragma unroll
      for (int v = 0; v < 8; ++v) {
        const int r = r0 + mt * 16 + v + hi * 8;
        const int j = n0 + nt * 16 + n;
        out[(size_t)r * D_ + j] = acc[mt][nt][v] + bo[j];
      }
    }
  }
}

// ---------------------------------------------------------------------------
// Host launcher
// ---------------------------------------------------------------------------
extern "C" void kernel_launch(void* const* d_in, const int* in_sizes, int n_in,
                              void* d_out, int out_size, void* d_ws,
                              size_t ws_size, hipStream_t stream) {
  const float* x  = (const float*)d_in[0];
  const float* wq = (const float*)d_in[1];
  const float* bq = (const float*)d_in[2];
  const float* wk = (const float*)d_in[3];
  const float* bk = (const float*)d_in[4];
  const float* wv = (const float*)d_in[5];
  const float* bv = (const float*)d_in[6];
  const float* wo = (const float*)d_in[7];
  const float* bo = (const float*)d_in[8];

  float* out  = (float*)d_out;
  float* attn = out + (size_t)B_ * S_ * D_;  // tuple: (out, attn) concatenated

  const size_t BSD = (size_t)B_ * S_ * D_;  // 8,388,608
  const size_t DD  = (size_t)D_ * D_;       // 1,048,576

  __bf16* xb  = (__bf16*)d_ws;
  __bf16* wqt = xb + BSD;
  __bf16* wkt = wqt + DD;
  __bf16* wvt = wkt + DD;
  __bf16* wot = wvt + DD;
  __bf16* Q   = wot + DD;
  __bf16* K   = Q + BSD;
  __bf16* Vt  = K + BSD;
  __bf16* ctx = Vt + BSD;  // total ws use ~92 MB bf16

  k_cvt_bf16<<<dim3((unsigned)(BSD / 256)), dim3(256), 0, stream>>>(x, xb,
                                                                    (int)BSD);
  k_transpose_cvt<<<dim3((unsigned)(DD / 256), 1, 4), dim3(256), 0, stream>>>(
      wq, wk, wv, wo, wqt, wkt, wvt, wot);
  k_qkv_gemm<<<dim3(256, 16, 3), dim3(32), 0, stream>>>(
      xb, wqt, wkt, wvt, bq, bk, bv, Q, K, Vt);
  k_attention<<<dim3(128, 16, 4), dim3(32), 0, stream>>>(Q, K, Vt, attn, ctx);
  k_out_gemm<<<dim3(256, 16), dim3(32), 0, stream>>>(ctx, wot, bo, out);
}
